// GAT_68590627717599
// MI455X (gfx1250) — compile-verified, matrix operand order checked by reading, not run on hardware
//
#include <hip/hip_runtime.h>
#include <hip/hip_bf16.h>

#define N       4096
#define NFEAT   256
#define NHID    64
#define NCLASS  40
#define NLAYERS 3
#define LS      72   // LDS row stride (bf16 elems): 144B = 9*16B, b128-aligned, spreads banks

typedef __bf16 bf16_t;
typedef __bf16 v16bf __attribute__((ext_vector_type(16)));
typedef float  v8f   __attribute__((ext_vector_type(8)));

// ---------------- helpers ----------------

__device__ __forceinline__ bf16_t f2bf(float f) {
  union { float f; unsigned u; } v; v.f = f;
  unsigned r = (v.u + 0x7FFFu + ((v.u >> 16) & 1u)) >> 16;   // RNE
  union { unsigned short s; bf16_t b; } o; o.s = (unsigned short)r;
  return o.b;
}

// pack two fp32 -> packed bf16x2 dword
__device__ __forceinline__ unsigned f2bf2(float a, float b) {
  union { float f; unsigned u; } x, y; x.f = a; y.f = b;
  unsigned lo = (x.u + 0x7FFFu + ((x.u >> 16) & 1u)) >> 16;
  unsigned hi = (y.u + 0x7FFFu + ((y.u >> 16) & 1u)) >> 16;
  return (lo & 0xFFFFu) | (hi << 16);
}

__device__ __forceinline__ v8f zero8() {
  v8f z;
#pragma unroll
  for (int e = 0; e < 8; ++e) z[e] = 0.0f;
  return z;
}

// Async global->LDS 16-byte copy (CDNA5 GLOBAL_LOAD_ASYNC_TO_LDS_B128, ASYNCcnt).
// LDS address = low 32 bits of the flat shared pointer (aperture truncation).
__device__ __forceinline__ void async_g2l_b128(void* lds_dst, const void* gsrc) {
  unsigned l = (unsigned)(uintptr_t)lds_dst;
  asm volatile("global_load_async_to_lds_b128 %0, %1, off"
               :: "v"(l), "v"(gsrc) : "memory");
}

__device__ __forceinline__ void wait_async0() {
  asm volatile("s_wait_asynccnt 0" ::: "memory");
}

// Load one 16x32 bf16 operand fragment (A layout; B uses same with row = n).
// Per ISA 7.12.2: elem j holds k = kk + (j/8)*16 + half*8 + (j%8); pairs are
// contiguous -> lowers to two ds_load_b128 per fragment.
__device__ __forceinline__ v16bf load_frag(const bf16_t* rowbase, int kk, int half) {
  union { v16bf v; unsigned u[8]; } r;
  const bf16_t* base = rowbase + kk + (half << 3);
#pragma unroll
  for (int q = 0; q < 8; ++q) {
    int k = ((q >> 2) << 4) + ((q & 3) << 1);
    r.u[q] = *(const unsigned*)(base + k);
  }
  return r.v;
}

__device__ __forceinline__ v8f wmma_bf16(v16bf a, v16bf b, v8f c) {
  return __builtin_amdgcn_wmma_f32_16x16x32_bf16(false, a, false, b,
                                                 (short)0, c, false, false);
}

// ---------------- fc1: h = x @ W1[i] + b1[i]; also hT (bf16) ----------------

__global__ __launch_bounds__(128) void fc1_kernel(
    const float* __restrict__ x,   // [N][NFEAT]
    const float* __restrict__ W1,  // [3][NFEAT][NHID]
    const float* __restrict__ b1,  // [3][NHID]
    float* __restrict__ h,         // [3][N][NHID]
    bf16_t* __restrict__ hT)       // [3][NHID][N]
{
  const int i    = blockIdx.y;
  const int rb   = blockIdx.x;
  const int tid  = threadIdx.x;
  const int lane = tid & 31;
  const int w    = tid >> 5;
  const int half = lane >> 4;
  const int l15  = lane & 15;

  __shared__ bf16_t ldsA[64 * LS];
  __shared__ bf16_t ldsBt[64 * LS];

  const float* W1i = W1 + (size_t)i * NFEAT * NHID;

  v8f acc[4];
#pragma unroll
  for (int nt = 0; nt < 4; ++nt) acc[nt] = zero8();

  for (int kb = 0; kb < NFEAT; kb += 64) {
    __syncthreads();
    for (int t = tid; t < 64 * 32; t += 128) {          // x tile, fp32 pairs -> bf16x2
      int r = t >> 5, c2 = (t & 31) * 2;
      float2 v = *(const float2*)(x + (size_t)(rb * 64 + r) * NFEAT + kb + c2);
      *(unsigned*)(ldsA + r * LS + c2) = f2bf2(v.x, v.y);
    }
    for (int t = tid; t < 64 * 64; t += 128) {          // W1 tile transposed
      int n = t >> 6, k = t & 63;
      ldsBt[n * LS + k] = f2bf(W1i[(size_t)(kb + k) * NHID + n]);
    }
    __syncthreads();
#pragma unroll
    for (int s = 0; s < 2; ++s) {
      v16bf a = load_frag(ldsA + (w * 16 + l15) * LS, s * 32, half);
#pragma unroll
      for (int nt = 0; nt < 4; ++nt) {
        v16bf b = load_frag(ldsBt + (nt * 16 + l15) * LS, s * 32, half);
        acc[nt] = wmma_bf16(a, b, acc[nt]);
      }
    }
  }

  float*  hi  = h  + (size_t)i * N * NHID;
  bf16_t* hTi = hT + (size_t)i * NHID * N;
#pragma unroll
  for (int nt = 0; nt < 4; ++nt) {
#pragma unroll
    for (int r = 0; r < 8; ++r) {
      int row = rb * 64 + w * 16 + (half << 3) + r;
      int col = nt * 16 + l15;
      float v = acc[nt][r] + b1[i * NHID + col];
      hi[(size_t)row * NHID + col] = v;
      hTi[(size_t)col * N + row]   = f2bf(v);
    }
  }
}

// ------- h_att = adj_att @ h[i]  (bf16 out); 2 M-tiles per wave -------

__global__ __launch_bounds__(128) void hatt_kernel(
    const float*  __restrict__ adj,  // [N][N] fp32
    const bf16_t* __restrict__ hTi,  // [NHID][N] bf16 (branch slice)
    bf16_t* __restrict__ hatt)       // [N][NHID] bf16
{
  const int rb   = blockIdx.x;       // 128-row block
  const int tid  = threadIdx.x;
  const int lane = tid & 31;
  const int w    = tid >> 5;
  const int half = lane >> 4;
  const int l15  = lane & 15;

  __shared__ bf16_t ldsA[128 * LS];
  __shared__ bf16_t ldsBt[64 * LS];

  v8f acc[2][4];
#pragma unroll
  for (int m = 0; m < 2; ++m)
#pragma unroll
    for (int nt = 0; nt < 4; ++nt) acc[m][nt] = zero8();

  for (int kb = 0; kb < N; kb += 64) {
    __syncthreads();
    // async copy: B tile (bf16, already [n][k] contiguous) -> LDS
    for (int t = tid; t < 512; t += 128) {
      int n = t >> 3, ch = (t & 7) * 8;                 // 8 bf16 = 16B chunks
      async_g2l_b128(ldsBt + n * LS + ch, hTi + (size_t)n * N + kb + ch);
    }
    // overlap: A tile fp32 -> bf16 conversion (128 rows x 64 cols, pairs)
    for (int t = tid; t < 128 * 32; t += 128) {
      int r = t >> 5, c2 = (t & 31) * 2;
      const float* src = adj + (size_t)(rb * 128 + r) * N + kb + c2;
      float2 v = *(const float2*)src;
      *(unsigned*)(ldsA + r * LS + c2) = f2bf2(v.x, v.y);
    }
    // prefetch next k-tile of the adj stream (128B grain)
    if (kb + 64 < N) {
      for (int t = tid; t < 256; t += 128) {
        int r = t >> 1;
        __builtin_prefetch(adj + (size_t)(rb * 128 + r) * N + kb + 64 + (t & 1) * 32, 0, 0);
      }
    }
    wait_async0();
    __syncthreads();
#pragma unroll
    for (int s = 0; s < 2; ++s) {
      v16bf a0 = load_frag(ldsA + (w * 32 + l15) * LS, s * 32, half);
      v16bf a1 = load_frag(ldsA + (w * 32 + 16 + l15) * LS, s * 32, half);
#pragma unroll
      for (int nt = 0; nt < 4; ++nt) {
        v16bf b = load_frag(ldsBt + (nt * 16 + l15) * LS, s * 32, half);
        acc[0][nt] = wmma_bf16(a0, b, acc[0][nt]);
        acc[1][nt] = wmma_bf16(a1, b, acc[1][nt]);
      }
    }
  }

#pragma unroll
  for (int m = 0; m < 2; ++m)
#pragma unroll
    for (int nt = 0; nt < 4; ++nt)
#pragma unroll
      for (int r = 0; r < 8; ++r) {
        int row = rb * 128 + w * 32 + m * 16 + (half << 3) + r;
        int col = nt * 16 + l15;
        hatt[(size_t)row * NHID + col] = f2bf(acc[m][nt][r]);
      }
}

// ---------------- Q = h_att@WQ+bQ ; K = h_att@WK+bK (bf16 out) ----------------

__global__ __launch_bounds__(128) void qk_kernel(
    const bf16_t* __restrict__ hatt,  // [N][NHID]
    const float* __restrict__ WQ, const float* __restrict__ bQ,
    const float* __restrict__ WK, const float* __restrict__ bK,
    bf16_t* __restrict__ Qo, bf16_t* __restrict__ Ko)
{
  const int rb   = blockIdx.x;
  const int tid  = threadIdx.x;
  const int lane = tid & 31;
  const int w    = tid >> 5;
  const int half = lane >> 4;
  const int l15  = lane & 15;

  __shared__ bf16_t ldsA[64 * LS];
  __shared__ bf16_t ldsWq[64 * LS];
  __shared__ bf16_t ldsWk[64 * LS];

  for (int t = tid; t < 512; t += 128) {               // h_att rows: async copy
    int r = t >> 3, ch = (t & 7) * 8;
    async_g2l_b128(ldsA + r * LS + ch, hatt + (size_t)(rb * 64 + r) * NHID + ch);
  }
  for (int t = tid; t < 64 * 64; t += 128) {           // weights transposed, fp32->bf16
    int n = t >> 6, k = t & 63;
    ldsWq[n * LS + k] = f2bf(WQ[(size_t)k * NHID + n]);
    ldsWk[n * LS + k] = f2bf(WK[(size_t)k * NHID + n]);
  }
  wait_async0();
  __syncthreads();

  v8f accq[4], acck[4];
#pragma unroll
  for (int nt = 0; nt < 4; ++nt) { accq[nt] = zero8(); acck[nt] = zero8(); }

#pragma unroll
  for (int s = 0; s < 2; ++s) {
    v16bf a = load_frag(ldsA + (w * 16 + l15) * LS, s * 32, half);
#pragma unroll
    for (int nt = 0; nt < 4; ++nt) {
      v16bf bq = load_frag(ldsWq + (nt * 16 + l15) * LS, s * 32, half);
      v16bf bk = load_frag(ldsWk + (nt * 16 + l15) * LS, s * 32, half);
      accq[nt] = wmma_bf16(a, bq, accq[nt]);
      acck[nt] = wmma_bf16(a, bk, acck[nt]);
    }
  }

#pragma unroll
  for (int nt = 0; nt < 4; ++nt)
#pragma unroll
    for (int r = 0; r < 8; ++r) {
      int row = rb * 64 + w * 16 + (half << 3) + r;
      int col = nt * 16 + l15;
      Qo[(size_t)row * NHID + col] = f2bf(accq[nt][r] + bQ[col]);
      Ko[(size_t)row * NHID + col] = f2bf(acck[nt][r] + bK[col]);
    }
}

// ------- fused: h'[i] = ((Q@K^T) .* adjs) @ h[i]; S never hits HBM -------
// 128 Q-rows per block, 2 M-tiles per wave.

__global__ __launch_bounds__(128) void attn_kernel(
    const bf16_t* __restrict__ Q,    // [N][NHID]
    const bf16_t* __restrict__ K,    // [N][NHID]
    const float*  __restrict__ adjm, // [N][N] mask (branch slice)
    const bf16_t* __restrict__ hTi,  // [NHID][N]
    float* __restrict__ hout)        // [N][NHID] overwrites h[i]
{
  const int rb   = blockIdx.x;       // 128-row block
  const int tid  = threadIdx.x;
  const int lane = tid & 31;
  const int w    = tid >> 5;
  const int half = lane >> 4;
  const int l15  = lane & 15;

  __shared__ bf16_t ldsK[64 * LS];
  __shared__ bf16_t ldsH[64 * LS];
  __shared__ bf16_t ldsS[128 * LS];

  // Q fragments: invariant over column blocks
  v16bf aq[2][2];
#pragma unroll
  for (int m = 0; m < 2; ++m) {
    const bf16_t* qrow = Q + (size_t)(rb * 128 + w * 32 + m * 16 + l15) * NHID;
#pragma unroll
    for (int s = 0; s < 2; ++s) aq[m][s] = load_frag(qrow, s * 32, half);
  }

  v8f accH[2][4];
#pragma unroll
  for (int m = 0; m < 2; ++m)
#pragma unroll
    for (int nt = 0; nt < 4; ++nt) accH[m][nt] = zero8();

  for (int cb = 0; cb < N / 64; ++cb) {
    __syncthreads();
    for (int t = tid; t < 512; t += 128) {             // K rows (bf16 [n][k]) async
      int n = t >> 3, ch = (t & 7) * 8;
      async_g2l_b128(ldsK + n * LS + ch, K + (size_t)(cb * 64 + n) * NHID + ch);
    }
    for (int t = tid; t < 512; t += 128) {             // hT[n][cb*64+k] async
      int n = t >> 3, ch = (t & 7) * 8;
      async_g2l_b128(ldsH + n * LS + ch, hTi + (size_t)n * N + cb * 64 + ch);
    }
    // prefetch this block's mask stream (128 rows x 256B, 128B grain)
    for (int t = tid; t < 256; t += 128) {
      int r = t >> 1;
      __builtin_prefetch(adjm + (size_t)(rb * 128 + r) * N + cb * 64 + (t & 1) * 32, 0, 0);
    }
    wait_async0();
    __syncthreads();

    // S = Q @ K^T
    v8f accS[2][4];
#pragma unroll
    for (int m = 0; m < 2; ++m)
#pragma unroll
      for (int nt = 0; nt < 4; ++nt) accS[m][nt] = zero8();
#pragma unroll
    for (int s = 0; s < 2; ++s) {
#pragma unroll
      for (int nt = 0; nt < 4; ++nt) {
        v16bf b = load_frag(ldsK + (nt * 16 + l15) * LS, s * 32, half);
        accS[0][nt] = wmma_bf16(aq[0][s], b, accS[0][nt]);
        accS[1][nt] = wmma_bf16(aq[1][s], b, accS[1][nt]);
      }
    }

    // mask in f32, requantize S -> bf16 into wave-private LDS rows
#pragma unroll
    for (int m = 0; m < 2; ++m)
#pragma unroll
      for (int nt = 0; nt < 4; ++nt)
#pragma unroll
        for (int r = 0; r < 8; ++r) {
          int rloc = w * 32 + m * 16 + (half << 3) + r;
          int row  = rb * 128 + rloc;
          int col  = cb * 64 + nt * 16 + l15;
          float sv = accS[m][nt][r] * adjm[(size_t)row * N + col];
          ldsS[rloc * LS + nt * 16 + l15] = f2bf(sv);
        }
    __syncthreads();

    // h' += S @ h
#pragma unroll
    for (int s = 0; s < 2; ++s) {
#pragma unroll
      for (int m = 0; m < 2; ++m) {
        v16bf as = load_frag(ldsS + (w * 32 + m * 16 + l15) * LS, s * 32, half);
#pragma unroll
        for (int nt = 0; nt < 4; ++nt) {
          v16bf b = load_frag(ldsH + (nt * 16 + l15) * LS, s * 32, half);
          accH[m][nt] = wmma_bf16(as, b, accH[m][nt]);
        }
      }
    }
  }

#pragma unroll
  for (int m = 0; m < 2; ++m)
#pragma unroll
    for (int nt = 0; nt < 4; ++nt)
#pragma unroll
      for (int r = 0; r < 8; ++r) {
        int row = rb * 128 + w * 32 + m * 16 + (half << 3) + r;
        int col = nt * 16 + l15;
        hout[(size_t)row * NHID + col] = accH[m][nt][r];
      }
}

// ------- head: softmax(att) mix, L2-normalize, relu, fc2, log_softmax -------

__global__ __launch_bounds__(64) void head_kernel(
    const float* __restrict__ h,    // [3][N][NHID]
    const float* __restrict__ att,  // [3]
    const float* __restrict__ W2,   // [192][40]
    const float* __restrict__ b2,   // [40]
    float* __restrict__ out)        // [N][NCLASS]
{
  __shared__ float fbuf[64][200];
  const int row = blockIdx.x * 64 + threadIdx.x;

  float a0 = att[0], a1 = att[1], a2 = att[2];
  float mx = fmaxf(a0, fmaxf(a1, a2));
  float e0 = __expf(a0 - mx), e1 = __expf(a1 - mx), e2 = __expf(a2 - mx);
  float inv = 1.0f / (e0 + e1 + e2);
  float mask[3] = { e0 * inv, e1 * inv, e2 * inv };

  for (int i = 0; i < NLAYERS; ++i) {
    const float* hr = h + ((size_t)i * N + row) * NHID;
    float ss = 0.0f;
    for (int j = 0; j < NHID; ++j) { float v = hr[j]; ss += v * v; }
    float sc = mask[i] / fmaxf(sqrtf(ss), 1e-12f);
    for (int j = 0; j < NHID; ++j)
      fbuf[threadIdx.x][i * NHID + j] = fmaxf(hr[j] * sc, 0.0f);
  }

  float z[NCLASS];
#pragma unroll
  for (int c = 0; c < NCLASS; ++c) z[c] = b2[c];
  for (int j = 0; j < NHID * NLAYERS; ++j) {
    float fv = fbuf[threadIdx.x][j];
#pragma unroll
    for (int c = 0; c < NCLASS; ++c) z[c] += fv * W2[(size_t)j * NCLASS + c];
  }

  float m = z[0];
#pragma unroll
  for (int c = 1; c < NCLASS; ++c) m = fmaxf(m, z[c]);
  float s = 0.0f;
#pragma unroll
  for (int c = 0; c < NCLASS; ++c) s += __expf(z[c] - m);
  float ls = __logf(s);
#pragma unroll
  for (int c = 0; c < NCLASS; ++c)
    out[(size_t)row * NCLASS + c] = z[c] - m - ls;
}

// ---------------- launcher ----------------

extern "C" void kernel_launch(void* const* d_in, const int* in_sizes, int n_in,
                              void* d_out, int out_size, void* d_ws, size_t ws_size,
                              hipStream_t stream) {
  (void)in_sizes; (void)n_in; (void)out_size; (void)ws_size;

  const float* x      = (const float*)d_in[0];
  const float* adjatt = (const float*)d_in[1];
  const float* adjs   = (const float*)d_in[2];
  const float* W1     = (const float*)d_in[3];
  const float* b1     = (const float*)d_in[4];
  const float* WQ     = (const float*)d_in[5];
  const float* bQ     = (const float*)d_in[6];
  const float* WK     = (const float*)d_in[7];
  const float* bK     = (const float*)d_in[8];
  const float* att    = (const float*)d_in[9];
  const float* W2     = (const float*)d_in[10];
  const float* b2     = (const float*)d_in[11];
  float* out = (float*)d_out;

  char* ws = (char*)d_ws;
  float*  h    = (float*)(ws + 0);                              // 3 MB
  bf16_t* hT   = (bf16_t*)(ws + 3145728);                       // 1.5 MB
  bf16_t* hatt = (bf16_t*)(ws + 3145728 + 1572864);             // 512 KB
  bf16_t* Qb   = (bf16_t*)(ws + 3145728 + 1572864 + 524288);    // 512 KB
  bf16_t* Kb   = (bf16_t*)(ws + 3145728 + 1572864 + 2*524288);  // 512 KB

  dim3 b128(128);
  fc1_kernel<<<dim3(N / 64, NLAYERS), b128, 0, stream>>>(x, W1, b1, h, hT);

  for (int i = 1; i < NLAYERS; ++i) {
    const bf16_t* hTi = hT + (size_t)i * NHID * N;
    hatt_kernel<<<dim3(N / 128), b128, 0, stream>>>(adjatt, hTi, hatt);
    qk_kernel<<<dim3(N / 64), b128, 0, stream>>>(
        hatt,
        WQ + (size_t)(i - 1) * NHID * NHID, bQ + (size_t)(i - 1) * NHID,
        WK + (size_t)(i - 1) * NHID * NHID, bK + (size_t)(i - 1) * NHID,
        Qb, Kb);
    attn_kernel<<<dim3(N / 128), b128, 0, stream>>>(
        Qb, Kb, adjs + (size_t)(i - 1) * N * N, hTi,
        h + (size_t)i * N * NHID);
  }

  head_kernel<<<dim3(N / 64), dim3(64), 0, stream>>>(h, att, W2, b2, out);
}
